// ObjectCondensationLoss_66967130079579
// MI455X (gfx1250) — compile-verified, matrix operand order checked by reading, not run on hardware
//
#include <hip/hip_runtime.h>
#include <hip/hip_bf16.h>
#include <math.h>

#define NB 2
#define NN 4096
#define ND 32
#define NSEG 64
#define THRESH 0.5f
#define MARG 0.2f
#define MARGIN_W 5.0f

#define NTILES (NN / 16)      // max 256 tiles per batch (worst case all cp)
#define CHUNKS 8              // j-tile strips per i-tile
#define JPC (NTILES / CHUNKS) // 32 j-tiles per strip
#define NWAVES (NB * NTILES * CHUNKS) // 4096 waves for repulsion (static worst-case grid)

typedef __attribute__((ext_vector_type(16))) _Float16 v16h;
typedef __attribute__((ext_vector_type(8)))  _Float16 v8h;
typedef __attribute__((ext_vector_type(8)))  float    v8f;

// ---------------- workspace layout (bytes) ----------------
static constexpr size_t O_HALF  = 0;                                  // f16 copy of embed
static constexpr size_t O_CE    = O_HALF  + (size_t)NB * NN * ND * 2; // compacted cp f16 rows
static constexpr size_t O_SQ    = O_CE    + (size_t)NB * NN * ND * 2; // ||e||^2 (all points)
static constexpr size_t O_SQC   = O_SQ    + (size_t)NB * NN * 4;      // ||e||^2 (compacted, padded)
static constexpr size_t O_LAB   = O_SQC   + (size_t)NB * NN * 4;      // cp labels (0/1)
static constexpr size_t O_FIRST = O_LAB   + (size_t)NB * NN * 4;      // first-cp per segment (int)
static constexpr size_t O_BCE   = O_FIRST + (size_t)NB * NSEG * 4;    // per-point bce terms
static constexpr size_t O_PM    = O_BCE   + (size_t)NB * NN * 4;      // per-point pos-margin
static constexpr size_t O_NM    = O_PM    + (size_t)NB * NN * 4;      // per-point neg-margin
static constexpr size_t O_D2    = O_NM    + (size_t)NB * NN * 4;      // per-point attraction d2
static constexpr size_t O_ATTR  = O_D2    + (size_t)NB * NN * 4;      // per-segment attraction
static constexpr size_t O_POS   = O_ATTR  + (size_t)NB * NSEG * 4;    // pos count per batch
static constexpr size_t O_PCNT  = O_POS   + 16;                       // compacted count per batch (int)
static constexpr size_t O_WAVE  = O_PCNT  + 16;                       // per-wave repulsion partials

__device__ __forceinline__ float softplusf(float x) {
    return fmaxf(x, 0.f) + log1pf(__expf(-fabsf(x)));
}

__device__ float block_sum(float v, float* sm) {
    const int t = threadIdx.x;
    sm[t] = v; __syncthreads();
    for (int s = 128; s > 0; s >>= 1) { if (t < s) sm[t] += sm[t + s]; __syncthreads(); }
    float r = sm[0]; __syncthreads();
    return r;
}

// ---------------- init: sentinel for segment-min ----------------
__global__ void init_kernel(int* __restrict__ first) {
    int t = threadIdx.x;
    if (t < NB * NSEG) first[t] = NN;
}

// ---------------- pass 1: labels, |e|^2, f16 convert, first-cp ----------------
__global__ void point1_kernel(const float* __restrict__ embed,
                              const int* __restrict__ is_cp,
                              const int* __restrict__ slice_id,
                              _Float16* __restrict__ halfE,
                              float* __restrict__ sq,
                              float* __restrict__ lab,
                              int* __restrict__ first) {
    const int idx = blockIdx.x * blockDim.x + threadIdx.x; // < NB*NN
    const int b = idx / NN, n = idx % NN;
    const int cp = is_cp[idx];
    lab[idx] = (cp == 1) ? 1.f : 0.f;
    const float* e = embed + (size_t)idx * ND;
    _Float16* h = halfE + (size_t)idx * ND;
    float s = 0.f;
    #pragma unroll
    for (int d = 0; d < ND; ++d) { float v = e[d]; s += v * v; h[d] = (_Float16)v; }
    sq[idx] = s;
    if (cp == 1) atomicMin(&first[b * NSEG + slice_id[idx]], n);
}

// ---------------- pos count per batch (deterministic tree) ----------------
__global__ void pos_kernel(const float* __restrict__ lab, float* __restrict__ posArr) {
    __shared__ float sm[256];
    const int b = blockIdx.x;
    float s = 0.f;
    for (int n = threadIdx.x; n < NN; n += 256) s += lab[b * NN + n];
    float r = block_sum(s, sm);
    if (threadIdx.x == 0) posArr[b] = r;
}

// ---------------- compact cp rows (deterministic scan, one block per batch) ----------
__global__ void compact_kernel(const _Float16* __restrict__ halfE,
                               const float* __restrict__ sq,
                               const int* __restrict__ is_cp,
                               _Float16* __restrict__ cE,
                               float* __restrict__ sqC,
                               int* __restrict__ pcnt) {
    __shared__ int scan[256];
    __shared__ int totalP;
    const int b = blockIdx.x;
    const int t = threadIdx.x;
    const int n0 = t * 16; // 256 threads * 16 = 4096
    int cnt = 0;
    #pragma unroll
    for (int i = 0; i < 16; ++i) cnt += (is_cp[b * NN + n0 + i] == 1) ? 1 : 0;
    scan[t] = cnt;
    __syncthreads();
    // Hillis-Steele inclusive scan (deterministic)
    for (int off = 1; off < 256; off <<= 1) {
        int v = (t >= off) ? scan[t - off] : 0;
        __syncthreads();
        scan[t] += v;
        __syncthreads();
    }
    int base = scan[t] - cnt; // exclusive prefix
    if (t == 255) totalP = scan[t];
    __syncthreads();
    const size_t eBase = (size_t)b * NN * ND;
    for (int i = 0; i < 16; ++i) {
        const int n = n0 + i;
        if (is_cp[b * NN + n] == 1) {
            const _Float16* s = halfE + eBase + (size_t)n * ND;
            _Float16* d = cE + eBase + (size_t)base * ND;
            #pragma unroll
            for (int dd = 0; dd < ND; ++dd) d[dd] = s[dd];
            sqC[b * NN + base] = sq[b * NN + n];
            ++base;
        }
    }
    __syncthreads();
    const int P = totalP;
    const int Ppad = ((P + 15) >> 4) << 4;
    if (t == 0) pcnt[b] = P;
    // pad rows: zero embedding, huge |e|^2 so exp(-d2) == 0 exactly
    for (int r = P + t; r < Ppad; r += 256) {
        _Float16* d = cE + eBase + (size_t)r * ND;
        #pragma unroll
        for (int dd = 0; dd < ND; ++dd) d[dd] = (_Float16)0.f;
        sqC[b * NN + r] = 1e30f;
    }
}

// ---------------- pass 2: bce/margin terms, attraction d2 ----------------
__global__ void point2_kernel(const float* __restrict__ beta,
                              const float* __restrict__ embed,
                              const float* __restrict__ lab,
                              const int* __restrict__ slice_id,
                              const int* __restrict__ first,
                              const float* __restrict__ posArr,
                              float* __restrict__ bceA, float* __restrict__ pmA,
                              float* __restrict__ nmA, float* __restrict__ d2A) {
    const int idx = blockIdx.x * blockDim.x + threadIdx.x;
    const int b = idx / NN;
    const float posb = posArr[b];
    const float negb = (float)NN - posb;
    const float pw = negb / (posb + 1e-6f);
    const float labv = lab[idx];
    const float x = beta[idx];
    bceA[idx] = pw * labv * softplusf(-x) + (1.f - labv) * softplusf(x);
    const float prob = 1.f / (1.f + __expf(-x));
    pmA[idx] = fmaxf(THRESH + MARG - prob, 0.f) * labv;
    nmA[idx] = fmaxf(prob - (THRESH - MARG), 0.f) * (1.f - labv);
    int f = first[b * NSEG + slice_id[idx]];
    if (f > NN - 1) f = NN - 1;
    const float* e = embed + (size_t)idx * ND;
    const float* c = embed + ((size_t)b * NN + f) * ND;
    float d2 = 0.f;
    #pragma unroll
    for (int d = 0; d < ND; ++d) { float df = e[d] - c[d]; d2 += df * df; }
    d2A[idx] = d2;
}

// ---------------- segment means (one thread per (b,seg), deterministic) ----------------
__global__ void seg_kernel(const int* __restrict__ slice_id,
                           const float* __restrict__ d2A,
                           const int* __restrict__ first,
                           float* __restrict__ attrk) {
    const int t = threadIdx.x;
    if (t >= NB * NSEG) return;
    const int b = t / NSEG, k = t % NSEG;
    const int* sidb = slice_id + b * NN;
    const float* d2b = d2A + b * NN;
    float cnt = 0.f, s = 0.f;
    for (int n = 0; n < NN; ++n)
        if (sidb[n] == k) { cnt += 1.f; s += d2b[n]; }
    attrk[t] = (first[t] < NN) ? s / fmaxf(cnt, 1.f) : 0.f;
}

// ---------------- repulsion: WMMA Gram tiles over compacted cp panel ----------------
// Symmetric: only tiles jt >= itile; off-diagonal tiles weighted x2.
// Static worst-case wave grid; waves beyond the live tile range write 0.
__global__ void rep_kernel(const _Float16* __restrict__ cE,
                           const float* __restrict__ sqC,
                           const int* __restrict__ pcnt,
                           float* __restrict__ wavePart) {
    const int gw = blockIdx.x * (blockDim.x >> 5) + (threadIdx.x >> 5);
    const int lane = threadIdx.x & 31;
    const int b     = gw >> 11;          // 2048 waves per batch
    const int rem   = gw & 2047;
    const int itile = rem >> 3;
    const int chunk = rem & 7;

    const int P = pcnt[b];
    const int T = (P + 15) >> 4;         // live 16-row tiles

    int jlo = chunk * JPC; if (jlo < itile) jlo = itile;   // symmetry: jt >= itile
    int jhi = (chunk + 1) * JPC; if (jhi > T) jhi = T;

    float acc = 0.f;
    if (itile < T && jlo < jhi) {
        const int group = lane >> 4;
        const int mrow  = lane & 15;
        const size_t base = (size_t)b * NN * ND;

        // A operand (16x32 f16): per-lane K-half packing per ISA layout
        const _Float16* rowA = cE + base + (size_t)(itile * 16 + mrow) * ND;
        v8h a_lo = *(const v8h*)(rowA + (group ? 8 : 0));
        v8h a_hi = *(const v8h*)(rowA + (group ? 24 : 16));
        v16h A = __builtin_shufflevector(a_lo, a_hi,
                                         0,1,2,3,4,5,6,7,8,9,10,11,12,13,14,15);

        // C/D layout: lane holds rows M = v + 8*group, column N = lane&15
        float sqI[8];
        #pragma unroll
        for (int v = 0; v < 8; ++v)
            sqI[v] = sqC[b * NN + itile * 16 + v + 8 * group];

        for (int jt = jlo; jt < jhi; ++jt) {
            const int j0 = jt * 16;
            // B operand = panel^T: same per-lane packing (layouts mirror M<->N)
            const _Float16* rowB = cE + base + (size_t)(j0 + mrow) * ND;
            v8h b_lo = *(const v8h*)(rowB + (group ? 8 : 0));
            v8h b_hi = *(const v8h*)(rowB + (group ? 24 : 16));
            v16h Bm = __builtin_shufflevector(b_lo, b_hi,
                                              0,1,2,3,4,5,6,7,8,9,10,11,12,13,14,15);
            v8f c = {};
            c = __builtin_amdgcn_wmma_f32_16x16x32_f16(
                    /*neg_a=*/false, A, /*neg_b=*/false, Bm,
                    /*c_mod=*/(short)0, c, /*reuse_a=*/false, /*reuse_b=*/false);

            const float sqJ = sqC[b * NN + j0 + mrow];
            float ts = 0.f;
            #pragma unroll
            for (int v = 0; v < 8; ++v) {
                float d2 = fmaxf(sqI[v] + sqJ - 2.f * c[v], 0.f);
                ts += __expf(-d2);   // pad rows: d2 ~ 1e30 -> contributes exactly 0
            }
            acc += (jt == itile) ? ts : 2.f * ts;
        }
    }
    // wave32 butterfly reduce (wave-uniform control flow above)
    #pragma unroll
    for (int off = 16; off > 0; off >>= 1)
        acc += __shfl_xor(acc, off, 32);
    if (lane == 0) wavePart[gw] = acc;
}

// ---------------- finalize: deterministic reductions + assembly ----------------
__global__ void fin_kernel(const float* __restrict__ bceA, const float* __restrict__ pmA,
                           const float* __restrict__ nmA, const float* __restrict__ attrk,
                           const float* __restrict__ wavePart, const float* __restrict__ posArr,
                           float* __restrict__ out, int out_size) {
    __shared__ float sm[256];
    const int t = threadIdx.x;
    float bceS[NB], pmS[NB], nmS[NB], atS[NB], repS[NB];
    for (int b = 0; b < NB; ++b) {
        float s;
        s = 0.f; for (int n = t; n < NN; n += 256) s += bceA[b * NN + n]; bceS[b] = block_sum(s, sm);
        s = 0.f; for (int n = t; n < NN; n += 256) s += pmA[b * NN + n];  pmS[b]  = block_sum(s, sm);
        s = 0.f; for (int n = t; n < NN; n += 256) s += nmA[b * NN + n];  nmS[b]  = block_sum(s, sm);
        s = (t < NSEG) ? attrk[b * NSEG + t] : 0.f;                      atS[b]  = block_sum(s, sm);
        s = 0.f; for (int w = t; w < NWAVES / NB; w += 256) s += wavePart[b * (NWAVES / NB) + w];
        repS[b] = block_sum(s, sm);
    }
    if (t == 0) {
        float fin = 0.f, aE = 0.f, bE = 0.f, rE = 0.f, cnt = 0.f;
        for (int b = 0; b < NB; ++b) {
            const float pos = posArr[b];
            const float neg = (float)NN - pos;
            const float valid = (pos >= 1.f && neg >= 1.f) ? 1.f : 0.f;
            const float beta_loss = bceS[b] / (float)NN +
                MARGIN_W * (pmS[b] / fmaxf(pos, 1.f) + nmS[b] / fmaxf(neg, 1.f));
            const float attr = atS[b];
            const float rep  = (pos > 1.f) ? repS[b] / fmaxf(pos * pos, 1.f) : 0.f;
            const float tot  = beta_loss + attr + rep;
            fin += tot * valid; aE += attr * valid; bE += beta_loss * valid;
            rE += rep * valid; cnt += valid;
        }
        const float den = fmaxf(cnt, 1.f);
        out[0] = fin / den;
        if (out_size >= 4) {  // jax pytree flattens dict keys sorted: attr, beta, repl
            out[1] = aE / den;
            out[2] = bE / den;
            out[3] = rE / den;
        }
    }
}

extern "C" void kernel_launch(void* const* d_in, const int* in_sizes, int n_in,
                              void* d_out, int out_size, void* d_ws, size_t ws_size,
                              hipStream_t stream) {
    const float* beta     = (const float*)d_in[0];
    const float* embed    = (const float*)d_in[1];
    const int*   slice_id = (const int*)d_in[2];
    const int*   is_cp    = (const int*)d_in[3];

    char* ws = (char*)d_ws;
    _Float16* halfE = (_Float16*)(ws + O_HALF);
    _Float16* cE    = (_Float16*)(ws + O_CE);
    float* sq    = (float*)(ws + O_SQ);
    float* sqC   = (float*)(ws + O_SQC);
    float* lab   = (float*)(ws + O_LAB);
    int*   first = (int*)  (ws + O_FIRST);
    float* bceA  = (float*)(ws + O_BCE);
    float* pmA   = (float*)(ws + O_PM);
    float* nmA   = (float*)(ws + O_NM);
    float* d2A   = (float*)(ws + O_D2);
    float* attrk = (float*)(ws + O_ATTR);
    float* posArr= (float*)(ws + O_POS);
    int*   pcnt  = (int*)  (ws + O_PCNT);
    float* waveP = (float*)(ws + O_WAVE);

    init_kernel   <<<1, 128, 0, stream>>>(first);
    point1_kernel <<<(NB * NN) / 256, 256, 0, stream>>>(embed, is_cp, slice_id,
                                                        halfE, sq, lab, first);
    pos_kernel    <<<NB, 256, 0, stream>>>(lab, posArr);
    compact_kernel<<<NB, 256, 0, stream>>>(halfE, sq, is_cp, cE, sqC, pcnt);
    point2_kernel <<<(NB * NN) / 256, 256, 0, stream>>>(beta, embed, lab, slice_id,
                                                        first, posArr, bceA, pmA, nmA, d2A);
    seg_kernel    <<<1, 128, 0, stream>>>(slice_id, d2A, first, attrk);
    rep_kernel    <<<NWAVES / 8, 256, 0, stream>>>(cE, sqC, pcnt, waveP);
    fin_kernel    <<<1, 256, 0, stream>>>(bceA, pmA, nmA, attrk, waveP, posArr,
                                          (float*)d_out, out_size);
}
